// CrissCrossAttention_103079215634
// MI455X (gfx1250) — matrix operand order repaired and checked by reading, not compiled, not run on hardware
//
#include <hip/hip_runtime.h>

#define BB 4
#define CC 256
#define C8 32
#define NN 4096   // 64*64

typedef __attribute__((ext_vector_type(16))) __bf16   v16bf;
typedef __attribute__((ext_vector_type(8)))  float    v8f;
typedef __attribute__((ext_vector_type(4)))  unsigned u32x4;
typedef __attribute__((ext_vector_type(4)))  float    f32x4;
typedef __attribute__((ext_vector_type(4)))  int      i32x4;

// address-space qualified int4 for the async-LDS builtin's pointer params
typedef __attribute__((address_space(1))) i32x4 gas_i32x4;   // global
typedef __attribute__((address_space(3))) i32x4 las_i32x4;   // LDS

union FragBF { v16bf v; u32x4 q[2]; __bf16 h[16]; };
union FragF  { v8f   v; float f[8]; };
union PkBF   { __bf16 h[2]; unsigned u; };

__device__ __forceinline__ v8f wmma_bf16(v16bf a, v16bf b, v8f c) {
    // D = A(16x32 bf16) * B(32x16 bf16) + C(16x16 f32)
    return __builtin_amdgcn_wmma_f32_16x16x32_bf16(false, a, false, b,
                                                   (short)0, c, false, false);
}

// ---- CDNA5 async global->LDS (ASYNCcnt) with compile-safe fallback ---------
#if __has_builtin(__builtin_amdgcn_global_load_async_to_lds_b128)
#define HAS_ASYNC_LDS 1
#endif

__device__ __forceinline__ void async_b128(__bf16* lds_dst, const __bf16* gsrc) {
#ifdef HAS_ASYNC_LDS
    __builtin_amdgcn_global_load_async_to_lds_b128(
        (gas_i32x4*)(void*)gsrc,
        (las_i32x4*)(void*)lds_dst,
        0, 0);
#else
    *(u32x4*)lds_dst = *(const u32x4*)gsrc;
#endif
}

__device__ __forceinline__ void wait_async0() {
#ifdef HAS_ASYNC_LDS
#if __has_builtin(__builtin_amdgcn_s_wait_asynccnt)
    __builtin_amdgcn_s_wait_asynccnt(0);
#else
    asm volatile("s_wait_asynccnt 0" ::: "memory");
#endif
#endif
}

// ---------------------------------------------------------------------------
// Kernel 1: 1x1-conv projections  Q[B,N,32], K^T[B,N,32], V[B,C,N]  (bf16 out)
// Combined weight rows: 0-31 = Wq, 32-63 = Wk, 64-319 = Wv.
// Block: 64 pixels (n) x all 320 output rows.  256 threads = 8 waves.
// ---------------------------------------------------------------------------
__global__ __launch_bounds__(256) void proj_kernel(
    const float* __restrict__ x,
    const float* __restrict__ Wq, const float* __restrict__ bq,
    const float* __restrict__ Wk, const float* __restrict__ bk,
    const float* __restrict__ Wv, const float* __restrict__ bv,
    unsigned short* __restrict__ q_us,
    unsigned short* __restrict__ kt_us,
    unsigned short* __restrict__ v_us)
{
    __shared__ __bf16 xT[64 * 264];   // [n][c], stride 264 (16B-aligned rows)

    const int tid = threadIdx.x;
    const int b   = blockIdx.x >> 6;
    const int n0  = (blockIdx.x & 63) * 64;

    // stage x tile transposed (n-major -> c-contiguous rows), f32 -> bf16
    for (int i = tid; i < 64 * 256; i += 256) {
        int n = i & 63;          // fastest -> coalesced global read
        int c = i >> 6;
        xT[n * 264 + c] = (__bf16)x[((b * CC + c) << 12) + n0 + n];
    }
    __syncthreads();

    __bf16* qb  = reinterpret_cast<__bf16*>(q_us);
    __bf16* ktb = reinterpret_cast<__bf16*>(kt_us);
    __bf16* vb  = reinterpret_cast<__bf16*>(v_us);

    const int wave = tid >> 5;
    const int l    = tid & 31;
    const int lh   = (l >> 4) & 1;   // half-wave select
    const int ll   = l & 15;

    for (int ot = wave; ot < 20; ot += 8) {          // 20 o-tiles of 16 rows
        const int ro = ot * 16 + ll;                 // this lane's weight row
        const float* wr;
        if (ro < 32)       wr = Wq + ro * CC;
        else if (ro < 64)  wr = Wk + (ro - 32) * CC;
        else               wr = Wv + (ro - 64) * CC;

        // A fragments: W rows, K = input channel (8 chunks of 32)
        FragBF A[8];
        #pragma unroll
        for (int kc = 0; kc < 8; ++kc) {
            const int c0 = kc * 32 + lh * 8;
            const f32x4 w0 = *(const f32x4*)(wr + c0);
            const f32x4 w1 = *(const f32x4*)(wr + c0 + 4);
            const f32x4 w2 = *(const f32x4*)(wr + c0 + 16);
            const f32x4 w3 = *(const f32x4*)(wr + c0 + 20);
            A[kc].h[0]=(__bf16)w0.x;  A[kc].h[1]=(__bf16)w0.y;
            A[kc].h[2]=(__bf16)w0.z;  A[kc].h[3]=(__bf16)w0.w;
            A[kc].h[4]=(__bf16)w1.x;  A[kc].h[5]=(__bf16)w1.y;
            A[kc].h[6]=(__bf16)w1.z;  A[kc].h[7]=(__bf16)w1.w;
            A[kc].h[8]=(__bf16)w2.x;  A[kc].h[9]=(__bf16)w2.y;
            A[kc].h[10]=(__bf16)w2.z; A[kc].h[11]=(__bf16)w2.w;
            A[kc].h[12]=(__bf16)w3.x; A[kc].h[13]=(__bf16)w3.y;
            A[kc].h[14]=(__bf16)w3.z; A[kc].h[15]=(__bf16)w3.w;
        }

        float biasr[8];
        #pragma unroll
        for (int r = 0; r < 8; ++r) {
            const int og = ot * 16 + r + 8 * lh;
            biasr[r] = (og < 32) ? bq[og] : (og < 64) ? bk[og - 32] : bv[og - 64];
        }

        #pragma unroll
        for (int nt = 0; nt < 4; ++nt) {             // 4 n-tiles of 16 pixels
            FragF acc;
            #pragma unroll
            for (int i = 0; i < 8; ++i) acc.f[i] = 0.f;

            #pragma unroll
            for (int kc = 0; kc < 8; ++kc) {
                FragBF Bf;
                const int nl = nt * 16 + ll;
                const int cb = kc * 32 + lh * 8;
                Bf.q[0] = *(const u32x4*)&xT[nl * 264 + cb];
                Bf.q[1] = *(const u32x4*)&xT[nl * 264 + cb + 16];
                acc.v = wmma_bf16(A[kc].v, Bf.v, acc.v);
            }

            const int n = n0 + nt * 16 + ll;         // C-frag: N = lane%16
            if (ot < 2) {                            // Q rows: pack 2x bf16
                #pragma unroll
                for (int r = 0; r < 8; r += 2) {
                    PkBF p;
                    p.h[0] = (__bf16)(acc.f[r]     + biasr[r]);
                    p.h[1] = (__bf16)(acc.f[r + 1] + biasr[r + 1]);
                    const int og = ot * 16 + r + 8 * lh;
                    *(unsigned*)&qb[(b * NN + n) * C8 + og] = p.u;
                }
            } else if (ot < 4) {                     // K rows: pack 2x bf16
                #pragma unroll
                for (int r = 0; r < 8; r += 2) {
                    PkBF p;
                    p.h[0] = (__bf16)(acc.f[r]     + biasr[r]);
                    p.h[1] = (__bf16)(acc.f[r + 1] + biasr[r + 1]);
                    const int og = (ot - 2) * 16 + r + 8 * lh;
                    *(unsigned*)&ktb[(b * NN + n) * C8 + og] = p.u;
                }
            } else {                                 // V rows: [c][n] layout
                #pragma unroll
                for (int r = 0; r < 8; ++r) {
                    const int cO = (ot - 4) * 16 + r + 8 * lh;
                    vb[((b * CC + cO) << 12) + n] = (__bf16)(acc.f[r] + biasr[r]);
                }
            }
        }
    }
}

// ---------------------------------------------------------------------------
// Kernel 2: flash attention.  Block = 4 waves = 64 query rows; 16 rows/wave.
// Double-buffered K^T/V tiles staged with async global->LDS; online softmax;
// O accumulated across 32 bf16 WMMAs per 64-column stage.
// ---------------------------------------------------------------------------
__global__ __launch_bounds__(128) void attn_kernel(
    const float* __restrict__ x,
    const unsigned short* __restrict__ q_us,
    const unsigned short* __restrict__ kt_us,
    const unsigned short* __restrict__ v_us,
    const float* __restrict__ gamma,
    float* __restrict__ out)
{
    __shared__ __bf16 kt_lds[2][64 * 40];   // [n_local][o]   stride 40 (16B rows)
    __shared__ __bf16 vt_lds[2][256 * 80];  // [c][n_local]   stride 80 (16B rows)
    __shared__ __bf16 p_lds[4][16 * 64];    // per-wave P tile [m][n_local]
    __shared__ float  ostage[4][64 * 16];   // per-wave output transpose buffer

    const int tid  = threadIdx.x;
    const int b    = blockIdx.x >> 6;
    const int row0 = (blockIdx.x & 63) * 64;
    const int wave = tid >> 5;
    const int l    = tid & 31;
    const int lh   = (l >> 4) & 1;
    const int ll   = l & 15;
    const int m0w  = row0 + wave * 16;      // this wave's 16 query rows

    const __bf16* qb  = (const __bf16*)q_us;
    const __bf16* ktb = (const __bf16*)kt_us;
    const __bf16* vb  = (const __bf16*)v_us;

    // Q fragment (16 rows x K=32) — lives in registers for the whole kernel
    FragBF Qf;
    {
        const __bf16* qr = qb + (b * NN + m0w + ll) * C8;
        Qf.q[0] = *(const u32x4*)(qr + lh * 8);
        Qf.q[1] = *(const u32x4*)(qr + lh * 8 + 16);
    }

    FragF O[16];                            // 16 x (16x16 f32) = 16 rows x 256 ch
    #pragma unroll
    for (int cg = 0; cg < 16; ++cg)
        #pragma unroll
        for (int r = 0; r < 8; ++r) O[cg].f[r] = 0.f;

    float rowmax[8], rowsum[8];
    #pragma unroll
    for (int r = 0; r < 8; ++r) { rowmax[r] = -1e30f; rowsum[r] = 0.f; }

    // cooperative tile stage: issue async loads for tile starting at k0
    auto stage = [&](int buf, int k0) {
        {   // K^T tile (64 x 32)
            const int n = tid >> 1, hf = tid & 1;
            const __bf16* src = ktb + (b * NN + k0 + n) * C8 + hf * 16;
            __bf16* dst = &kt_lds[buf][n * 40 + hf * 16];
            async_b128(dst, src);
            async_b128(dst + 8, src + 8);
        }
        #pragma unroll
        for (int i = 0; i < 8; ++i) {   // V tile (256 x 64)
            const int c = (tid >> 2) + i * 32, nh = tid & 3;
            const __bf16* src = vb + ((b * CC + c) << 12) + k0 + nh * 16;
            __bf16* dst = &vt_lds[buf][c * 80 + nh * 16];
            async_b128(dst, src);
            async_b128(dst + 8, src + 8);
        }
    };

    stage(0, 0);                            // prologue

    for (int ki = 0; ki < NN / 64; ++ki) {
        const int cur = ki & 1;
        wait_async0();                      // my async loads into LDS done
        __syncthreads();                    // everyone's are done
        if (ki + 1 < NN / 64) stage(cur ^ 1, (ki + 1) * 64);   // prefetch next

        const __bf16* ktL = kt_lds[cur];
        const __bf16* vtL = vt_lds[cur];

        // ---- S = Q K^T for 4 column tiles of 16 ----
        FragF s[4];
        #pragma unroll
        for (int nt = 0; nt < 4; ++nt) {
            FragBF Kf;
            const __bf16* kr = &ktL[(nt * 16 + ll) * 40 + lh * 8];
            Kf.q[0] = ((const u32x4*)kr)[0];
            Kf.q[1] = *(const u32x4*)(kr + 16);
            v8f z = {0.f,0.f,0.f,0.f,0.f,0.f,0.f,0.f};
            s[nt].v = wmma_bf16(Qf.v, Kf.v, z);
        }

        // ---- online softmax (rows live across 16-lane groups) ----
        float sc[8];
        #pragma unroll
        for (int r = 0; r < 8; ++r) {
            float t = fmaxf(fmaxf(s[0].f[r], s[1].f[r]),
                            fmaxf(s[2].f[r], s[3].f[r]));
            t = fmaxf(t, __shfl_xor(t, 8, 16));
            t = fmaxf(t, __shfl_xor(t, 4, 16));
            t = fmaxf(t, __shfl_xor(t, 2, 16));
            t = fmaxf(t, __shfl_xor(t, 1, 16));
            const float nm    = fmaxf(rowmax[r], t);
            const float scale = __expf(rowmax[r] - nm);
            rowmax[r] = nm;
            float ls = 0.f;
            #pragma unroll
            for (int nt = 0; nt < 4; ++nt) {
                const float p = __expf(s[nt].f[r] - nm);
                s[nt].f[r] = p;
                ls += p;
            }
            rowsum[r] = rowsum[r] * scale + ls;
            sc[r] = scale;
        }
        #pragma unroll
        for (int cg = 0; cg < 16; ++cg)
            #pragma unroll
            for (int r = 0; r < 8; ++r) O[cg].f[r] *= sc[r];

        // ---- restage P: C-layout -> A-layout via per-wave LDS ----
        #pragma unroll
        for (int nt = 0; nt < 4; ++nt)
            #pragma unroll
            for (int r = 0; r < 8; ++r)
                p_lds[wave][(r + 8 * lh) * 64 + nt * 16 + ll] = (__bf16)s[nt].f[r];
        asm volatile("s_wait_dscnt 0" ::: "memory");

        // ---- O += P V^T : 2 K-chunks of 32 x 16 channel groups ----
        #pragma unroll
        for (int kt = 0; kt < 2; ++kt) {
            FragBF Pf;
            const __bf16* pr = &p_lds[wave][ll * 64 + kt * 32 + lh * 8];
            Pf.q[0] = ((const u32x4*)pr)[0];
            Pf.q[1] = *(const u32x4*)(pr + 16);
            #pragma unroll
            for (int cg = 0; cg < 16; ++cg) {
                FragBF Vf;
                const __bf16* vr = &vtL[(cg * 16 + ll) * 80 + kt * 32 + lh * 8];
                Vf.q[0] = ((const u32x4*)vr)[0];
                Vf.q[1] = *(const u32x4*)(vr + 16);
                O[cg].v = wmma_bf16(Pf.v, Vf.v, O[cg].v);
            }
        }
    }

    // ---- finalize: divide by rowsum, transpose via LDS, write gamma*O + x ----
    float inv[8];
    #pragma unroll
    for (int r = 0; r < 8; ++r) {
        float rs = rowsum[r];
        rs += __shfl_xor(rs, 8, 16);
        rs += __shfl_xor(rs, 4, 16);
        rs += __shfl_xor(rs, 2, 16);
        rs += __shfl_xor(rs, 1, 16);
        inv[r] = 1.f / rs;
    }
    const float g = gamma[0];

    for (int pass = 0; pass < 4; ++pass) {          // 64 channels per pass
        #pragma unroll
        for (int cgl = 0; cgl < 4; ++cgl) {
            const int cg = pass * 4 + cgl;
            #pragma unroll
            for (int r = 0; r < 8; ++r)
                ostage[wave][(cgl * 16 + ll) * 16 + r + 8 * lh] = O[cg].f[r] * inv[r];
        }
        asm volatile("s_wait_dscnt 0" ::: "memory");
        #pragma unroll
        for (int j = 0; j < 8; ++j) {
            const int c_loc = j * 8 + (l >> 2);
            const int c     = pass * 64 + c_loc;
            const int m     = m0w + (l & 3) * 4;
            const f32x4 ov  = *(const f32x4*)&ostage[wave][c_loc * 16 + (l & 3) * 4];
            const int gi    = ((b * CC + c) << 12) + m;
            const f32x4 xv  = *(const f32x4*)(x + gi);
            f32x4 o4;
            o4.x = g * ov.x + xv.x;  o4.y = g * ov.y + xv.y;
            o4.z = g * ov.z + xv.z;  o4.w = g * ov.w + xv.w;
            *(f32x4*)(out + gi) = o4;
        }
        asm volatile("s_wait_dscnt 0" ::: "memory");
    }
}

// ---------------------------------------------------------------------------
extern "C" void kernel_launch(void* const* d_in, const int* in_sizes, int n_in,
                              void* d_out, int out_size, void* d_ws, size_t ws_size,
                              hipStream_t stream)
{
    (void)in_sizes; (void)n_in; (void)out_size; (void)ws_size;
    const float* x     = (const float*)d_in[0];
    const float* Wq    = (const float*)d_in[1];
    const float* bq    = (const float*)d_in[2];
    const float* Wk    = (const float*)d_in[3];
    const float* bk    = (const float*)d_in[4];
    const float* Wv    = (const float*)d_in[5];
    const float* bv    = (const float*)d_in[6];
    const float* gamma = (const float*)d_in[7];
    float* out = (float*)d_out;

    char* ws = (char*)d_ws;
    unsigned short* qbf  = (unsigned short*)(ws);                 // 1 MB  [B,N,32]
    unsigned short* ktbf = (unsigned short*)(ws + (1u << 20));    // 1 MB  [B,N,32]
    unsigned short* vbf  = (unsigned short*)(ws + (2u << 20));    // 8 MB  [B,C,N]

    proj_kernel<<<dim3(BB * (NN / 64)), dim3(256), 0, stream>>>(
        x, Wq, bq, Wk, bk, Wv, bv, qbf, ktbf, vbf);
    attn_kernel<<<dim3(BB * (NN / 64)), dim3(128), 0, stream>>>(
        x, qbf, ktbf, vbf, gamma, out);
}
